// Coma_12086037971037
// MI455X (gfx1250) — matrix operand order repaired and checked by reading, not compiled, not run on hardware
//
#include <hip/hip_runtime.h>
#include <cstddef>

typedef __attribute__((ext_vector_type(16))) _Float16 v16h;
typedef __attribute__((ext_vector_type(8)))  float    v8f;

// ---------------------------------------------------------------------------
// Fully shape/flag-specialized WMMA GEMM: out[M,ND] = op( A[M,KD] x W + bias ).
//   TRANS : W is [ND,KD] row-major (computes A @ W^T); else W is [KD,ND]
//   BETA  : accumulate into existing out
//   BIAS  : add bias[ncol]
//   RELU  : clamp at 0 in the epilogue
// One wave per 16x16 C tile; K loop in steps of 32, zero padded. All template
// parameters constant-fold the padding/epilogue paths; the only lane-level
// padding left (KD==3 / ND==3) is clamped-address loads + selects — no EXEC
// divergence, all loads batch under a single wait.
// ---------------------------------------------------------------------------
template <int KD, int ND, bool TRANS, bool BETA, bool BIAS, bool RELU>
__global__ void wmma_gemm_kernel(const float* __restrict__ A,
                                 const float* __restrict__ W,
                                 const float* __restrict__ bias,
                                 float* __restrict__ out,
                                 int M)
{
    constexpr int Nt = (ND + 15) / 16;
    constexpr int Kt = (KD + 31) / 32;

    const int gwave = (blockIdx.x * blockDim.x + threadIdx.x) >> 5;
    const int lane  = threadIdx.x & 31;
    const int Mt = M >> 4;
    if (gwave >= Mt * Nt) return;          // wave-uniform: EXEC stays all-ones
    const int mt = gwave / Nt;
    const int nt = gwave % Nt;

    // A fragment (16x32 f16): lanes 0-15 -> M rows, K 0..7 & 16..23;
    // lanes 16-31 -> same rows, K 8..15 & 24..31.
    const int arow  = (mt << 4) + (lane & 15);
    const int kbase = (lane & 16) ? 8 : 0;
    const float* __restrict__ Arow = A + (size_t)arow * KD;
    const int n0 = nt << 4;

    v8f acc = {0.f, 0.f, 0.f, 0.f, 0.f, 0.f, 0.f, 0.f};
#pragma unroll
    for (int kt = 0; kt < Kt; ++kt) {
        const int k0 = kt << 5;
        v16h a, bfr;

        // ---------------- A fragment: two contiguous 8-wide K runs ----------
#pragma unroll
        for (int g2 = 0; g2 < 2; ++g2) {
            const int kg = k0 + (g2 << 4);            // compile-time after unroll
            if (kg + 16 <= KD) {                      // fully valid -> b128 loads
                const float4 f0 = *(const float4*)(Arow + kg + kbase);
                const float4 f1 = *(const float4*)(Arow + kg + kbase + 4);
                a[g2 * 8 + 0] = (_Float16)f0.x; a[g2 * 8 + 1] = (_Float16)f0.y;
                a[g2 * 8 + 2] = (_Float16)f0.z; a[g2 * 8 + 3] = (_Float16)f0.w;
                a[g2 * 8 + 4] = (_Float16)f1.x; a[g2 * 8 + 5] = (_Float16)f1.y;
                a[g2 * 8 + 6] = (_Float16)f1.z; a[g2 * 8 + 7] = (_Float16)f1.w;
            } else if (kg >= KD) {                    // fully out-of-range: zeros
#pragma unroll
                for (int j = 0; j < 8; ++j) a[g2 * 8 + j] = (_Float16)0.f;
            } else {                                  // partial (KD==3 only)
#pragma unroll
                for (int j = 0; j < 8; ++j) {
                    const int   kk = kg + kbase + j;
                    const float v  = Arow[kk < KD ? kk : 0];   // clamped load
                    a[g2 * 8 + j] = (kk < KD) ? (_Float16)v : (_Float16)0.f;
                }
            }
        }

        // ---------------- B fragment: lane%16 = N column, elements = K (same order as A) ------
        const int bn = n0 + (lane & 15);
        if (!TRANS && (n0 + 16 <= ND)) {
            // Clamp row, zero elements for OOR K rows.
#pragma unroll
            for (int s = 0; s < 16; ++s) {
                const int kk = k0 + kbase + (s & 7) + ((s & 8) << 1);
                const int cr = kk < KD ? kk : 0;
                const float wv = W[(size_t)cr * ND + bn];
                bfr[s] = (kk < KD) ? (_Float16)wv : (_Float16)0.f;
            }
        } else if (TRANS && (n0 + 16 <= ND) && (KD % 32 == 0)) {
            // Latent layers (K multiple of 32, N multiple of 16): unconditional.
#pragma unroll
            for (int s = 0; s < 16; ++s) {
                const int kk = k0 + kbase + (s & 7) + ((s & 8) << 1);
                bfr[s] = (_Float16)W[(size_t)bn * KD + kk];
            }
        } else {
            // Fallback (ND==3 tail): clamp + select.
            const int cn = bn < ND ? bn : 0;
#pragma unroll
            for (int s = 0; s < 16; ++s) {
                const int kk = k0 + kbase + (s & 7) + ((s & 8) << 1);
                const int cr = kk < KD ? kk : 0;
                const float wv = TRANS ? W[(size_t)cn * KD + cr]
                                       : W[(size_t)cr * ND + cn];
                bfr[s] = (kk < KD && bn < ND) ? (_Float16)wv : (_Float16)0.f;
            }
        }

        acc = __builtin_amdgcn_wmma_f32_16x16x32_f16(
                  false, a, false, bfr, (short)0, acc, false, false);
    }

    // C/D layout: VGPR i -> row (i + 8*(lane>=16)), col = lane%16.
    const int ncol  = n0 + (lane & 15);
    const int mrow0 = (mt << 4) + ((lane & 16) ? 8 : 0);
    if (ncol < ND) {
        float* __restrict__ base = out + (size_t)mrow0 * ND + ncol;
        float bv = 0.f;
        if (BIAS) bv = bias[ncol];
        float prev[8];
        if (BETA) {
#pragma unroll
            for (int i = 0; i < 8; ++i) prev[i] = base[(size_t)i * ND];
        }
#pragma unroll
        for (int i = 0; i < 8; ++i) {
            float v = acc[i];
            if (BIAS) v += bv;
            if (BETA) v += prev[i];
            if (RELU) v = fmaxf(v, 0.f);
            base[(size_t)i * ND] = v;
        }
    }
}

// ---------------------------------------------------------------------------
// Chebyshev propagation (deterministic gather form of the scatter-add).
// Edge list is structured: edge e = g*n_e + i has row == i (ring graph).
// Nodes i >= n_e receive no messages (final-conv case: coarse edges applied
// to the fine tensor): prop == 0 there, so T_k = -T_{k-2}.
//   prev2 == nullptr : out = prop(x)            (T1)
//   prev2 != nullptr : out = 2*prop(x) - prev2  (Tk, k>=2)
// ---------------------------------------------------------------------------
__global__ void prop_kernel(const float* __restrict__ x,
                            const int* __restrict__ col,
                            const float* __restrict__ nrm,
                            const float* __restrict__ prev2,
                            float* __restrict__ out,
                            int Bn, int n, int n_e, int groups, int F)
{
    const long t = (long)blockIdx.x * blockDim.x + threadIdx.x;
    const long total = (long)Bn * n * F;
    if (t >= total) return;
    const int f = (int)(t % F);
    const int i = (int)((t / F) % n);
    const int b = (int)(t / ((long)F * n));
    float acc = 0.f;
    if (i < n_e) {
        const long xb = (long)b * n * F;
        for (int g = 0; g < groups; ++g) {
            const int e = g * n_e + i;
            acc += nrm[e] * x[xb + (long)col[e] * F + f];
        }
    }
    out[t] = prev2 ? (2.f * acc - prev2[t]) : acc;
}

// Pool / unpool: out[b,i,f] = val[i] * x[b, col[i], f]   (row = arange(nout))
__global__ void pool_kernel(const float* __restrict__ x,
                            const int* __restrict__ idx,
                            const float* __restrict__ val,
                            float* __restrict__ out,
                            int Bn, int nin, int nout, int F)
{
    const long t = (long)blockIdx.x * blockDim.x + threadIdx.x;
    const long total = (long)Bn * nout * F;
    if (t >= total) return;
    const int f = (int)(t % F);
    const int i = (int)((t / F) % nout);
    const int b = (int)(t / ((long)F * nout));
    const int c = idx[nout + i];
    out[t] = val[i] * x[((long)b * nin + c) * F + f];
}

// deg/dinv for ChebConv norm (structured rows: deg[i] = #groups with row==i)
__global__ void dinv_kernel(const int* __restrict__ row,
                            float* __restrict__ dinv, int n, int groups)
{
    const int i = blockIdx.x * blockDim.x + threadIdx.x;
    if (i >= n) return;
    float cnt = 0.f;
    for (int g = 0; g < groups; ++g) cnt += (row[g * n + i] == i) ? 1.f : 0.f;
    dinv[i] = (cnt > 0.f) ? rsqrtf(cnt) : 0.f;
}

__global__ void norm_kernel(const int* __restrict__ ei,
                            const float* __restrict__ dinv,
                            float* __restrict__ nrm, int E)
{
    const int e = blockIdx.x * blockDim.x + threadIdx.x;
    if (e >= E) return;
    nrm[e] = dinv[ei[e]] * dinv[ei[E + e]];
}

// ---------------------------------------------------------------------------
// Host-side driver helpers
// ---------------------------------------------------------------------------
template <int KD, int ND, bool TRANS, bool BETA, bool BIAS, bool RELU>
static inline void launch_gemm(const float* A, const float* W, const float* bias,
                               float* out, int M, hipStream_t s)
{
    constexpr int Nt = (ND + 15) / 16;
    const int waves  = (M >> 4) * Nt;
    const int blocks = (waves + 7) / 8;
    wmma_gemm_kernel<KD, ND, TRANS, BETA, BIAS, RELU>
        <<<blocks, 256, 0, s>>>(A, W, bias, out, M);
}

static inline void launch_prop(const float* x, const int* col, const float* nrm,
                               const float* prev2, float* out,
                               int Bn, int n, int n_e, int g, int F, hipStream_t s)
{
    const long total = (long)Bn * n * F;
    const int blocks = (int)((total + 255) / 256);
    prop_kernel<<<blocks, 256, 0, s>>>(x, col, nrm, prev2, out, Bn, n, n_e, g, F);
}

static inline void launch_pool(const float* x, const int* idx, const float* val,
                               float* out, int Bn, int nin, int nout, int F,
                               hipStream_t s)
{
    const long total = (long)Bn * nout * F;
    const int blocks = (int)((total + 255) / 256);
    pool_kernel<<<blocks, 256, 0, s>>>(x, idx, val, out, Bn, nin, nout, F);
}

// Full Chebyshev conv (K=6): out = sum_k T_k @ W[k] (+bias)(+relu).
// Rotation: T1->R1, T2->R2, T3->R0, T4->R1, T5->R2. R0 may alias X (X is
// dead once T2 is computed); for X==d_in / X==OUTB, R0 is simply free.
template <int FIN, int FOUT, bool BIAS, bool RELU>
static void cheb_conv(const float* X, const float* W6, const float* bias,
                      float* outp, const int* ei_, const float* nrm,
                      int n, int n_e, int E_,
                      float* R0, float* R1, float* R2, hipStream_t s)
{
    const int Bn = 128;
    const int M  = Bn * n;
    const int g  = E_ / n_e;
    const int* col = ei_ + E_;
    constexpr int FF = FIN * FOUT;

    launch_gemm<FIN, FOUT, false, false, BIAS, false>(X,  W6 + 0 * FF, bias, outp, M, s);
    launch_prop(X,  col, nrm, nullptr, R1, Bn, n, n_e, g, FIN, s);
    launch_gemm<FIN, FOUT, false, true, false, false>(R1, W6 + 1 * FF, nullptr, outp, M, s);
    launch_prop(R1, col, nrm, X,       R2, Bn, n, n_e, g, FIN, s);
    launch_gemm<FIN, FOUT, false, true, false, false>(R2, W6 + 2 * FF, nullptr, outp, M, s);
    launch_prop(R2, col, nrm, R1,      R0, Bn, n, n_e, g, FIN, s);
    launch_gemm<FIN, FOUT, false, true, false, false>(R0, W6 + 3 * FF, nullptr, outp, M, s);
    launch_prop(R0, col, nrm, R2,      R1, Bn, n, n_e, g, FIN, s);
    launch_gemm<FIN, FOUT, false, true, false, false>(R1, W6 + 4 * FF, nullptr, outp, M, s);
    launch_prop(R1, col, nrm, R0,      R2, Bn, n, n_e, g, FIN, s);
    launch_gemm<FIN, FOUT, false, true, false, RELU>(R2, W6 + 5 * FF, nullptr, outp, M, s);
}

extern "C" void kernel_launch(void* const* d_in, const int* in_sizes, int n_in,
                              void* d_out, int out_size, void* d_ws, size_t ws_size,
                              hipStream_t stream)
{
    (void)n_in; (void)out_size; (void)ws_size;
    static const int NS[5] = {5023, 1256, 314, 79, 20};
    const int Bn = 128;

    const float* data = (const float*)d_in[0];
    const int* ei[5]; int E[5];
    for (int l = 0; l < 5; ++l) { ei[l] = (const int*)d_in[1 + l]; E[l] = in_sizes[1 + l] / 2; }
    const int* d_idx[4]; const float* d_val[4];
    const int* u_idx[4]; const float* u_val[4];
    for (int i = 0; i < 4; ++i) {
        d_idx[i] = (const int*)  d_in[6 + 4 * i + 0];
        d_val[i] = (const float*)d_in[6 + 4 * i + 1];
        u_idx[i] = (const int*)  d_in[6 + 4 * i + 2];
        u_val[i] = (const float*)d_in[6 + 4 * i + 3];
    }
    const float* We[4]; const float* be[4];
    for (int i = 0; i < 4; ++i) {
        We[i] = (const float*)d_in[22 + 2 * i];
        be[i] = (const float*)d_in[23 + 2 * i];
    }
    const float* Wd[5]; const float* bd[4];
    for (int i = 0; i < 4; ++i) {
        Wd[i] = (const float*)d_in[30 + 2 * i];
        bd[i] = (const float*)d_in[31 + 2 * i];
    }
    Wd[4] = (const float*)d_in[38];
    const float* enc_w = (const float*)d_in[39];
    const float* enc_b = (const float*)d_in[40];
    const float* dec_w = (const float*)d_in[41];
    const float* dec_b = (const float*)d_in[42];

    // Workspace layout: 4 big f32 buffers (B*5023*16 elems) + small region.
    const size_t BIGN = (size_t)Bn * 5023 * 16;
    float* R0   = (float*)d_ws;
    float* R1   = R0 + BIGN;
    float* R2   = R1 + BIGN;
    float* OUTB = R2 + BIGN;
    float* sm   = OUTB + BIGN;
    float* normp[5];
    for (int l = 0; l < 5; ++l) { normp[l] = sm; sm += E[l]; }
    float* dinvp = sm; sm += 5023;
    float* zbuf  = sm; sm += 128 * 64;
    float* dbuf  = sm; sm += 128 * 640;

    // Edge normalisation per level (stream-ordered, dinv scratch reused).
    for (int l = 0; l < 5; ++l) {
        const int n = NS[l], g = E[l] / n;
        dinv_kernel<<<(n + 255) / 256, 256, 0, stream>>>(ei[l], dinvp, n, g);
        norm_kernel<<<(E[l] + 255) / 256, 256, 0, stream>>>(ei[l], dinvp, normp[l], E[l]);
    }

    // ---------------- Encoder ----------------
    cheb_conv<3, 16, true, true>(data, We[0], be[0], OUTB, ei[0], normp[0],
                                 NS[0], NS[0], E[0], R0, R1, R2, stream);
    launch_pool(OUTB, d_idx[0], d_val[0], R0, Bn, NS[0], NS[1], 16, stream);
    cheb_conv<16, 16, true, true>(R0, We[1], be[1], OUTB, ei[1], normp[1],
                                  NS[1], NS[1], E[1], R0, R1, R2, stream);
    launch_pool(OUTB, d_idx[1], d_val[1], R0, Bn, NS[1], NS[2], 16, stream);
    cheb_conv<16, 16, true, true>(R0, We[2], be[2], OUTB, ei[2], normp[2],
                                  NS[2], NS[2], E[2], R0, R1, R2, stream);
    launch_pool(OUTB, d_idx[2], d_val[2], R0, Bn, NS[2], NS[3], 16, stream);
    cheb_conv<16, 32, true, true>(R0, We[3], be[3], OUTB, ei[3], normp[3],
                                  NS[3], NS[3], E[3], R0, R1, R2, stream);
    launch_pool(OUTB, d_idx[3], d_val[3], R0, Bn, NS[3], NS[4], 32, stream);

    // ---------------- Latent ----------------
    // z = x(128,640) @ enc_w^T + enc_b ; x_dec = relu(z @ declin_w^T + declin_b)
    launch_gemm<640, 64, true, false, true, false>(R0,   enc_w, enc_b, zbuf, 128, stream);
    launch_gemm<64, 640, true, false, true, true >(zbuf, dec_w, dec_b, dbuf, 128, stream);

    // ---------------- Decoder ----------------
    launch_pool(dbuf, u_idx[3], u_val[3], R0, Bn, NS[4], NS[3], 32, stream);
    cheb_conv<32, 16, true, true>(R0, Wd[0], bd[0], OUTB, ei[3], normp[3],
                                  NS[3], NS[3], E[3], R0, R1, R2, stream);
    launch_pool(OUTB, u_idx[2], u_val[2], R0, Bn, NS[3], NS[2], 16, stream);
    cheb_conv<16, 16, true, true>(R0, Wd[1], bd[1], OUTB, ei[2], normp[2],
                                  NS[2], NS[2], E[2], R0, R1, R2, stream);
    launch_pool(OUTB, u_idx[1], u_val[1], R0, Bn, NS[2], NS[1], 16, stream);
    cheb_conv<16, 16, true, true>(R0, Wd[2], bd[2], OUTB, ei[1], normp[1],
                                  NS[1], NS[1], E[1], R0, R1, R2, stream);
    launch_pool(OUTB, u_idx[0], u_val[0], R0, Bn, NS[1], NS[0], 16, stream);
    cheb_conv<16, 16, true, true>(R0, Wd[3], bd[3], OUTB, ei[0], normp[0],
                                  NS[0], NS[0], E[0], R0, R1, R2, stream);

    // Final conv: coarse (20-node) edge set applied to the 5023-node tensor,
    // no bias, no relu, writes d_out directly ([B*5023, 3] row-major).
    cheb_conv<16, 3, false, false>(OUTB, Wd[4], nullptr, (float*)d_out, ei[4], normp[4],
                                   NS[0], NS[4], E[4], R0, R1, R2, stream);
}
